// NMSLatentDisentangledDynamic_23115513987174
// MI455X (gfx1250) — matrix operand order repaired and checked by reading, not compiled, not run on hardware
//
#include <hip/hip_runtime.h>
#include <math.h>

// ---------------------------------------------------------------------------
// NMS latent-disentangled VAE pipeline for MI455X (gfx1250, wave32, WMMA).
//   encoder 1x1 conv -> BiLSTM -> 4 linear heads -> reparam -> GMM posterior
//   decoder BiLSTM x2 -> relu linear -> sigmoid linear
// All GEMMs run through v_wmma_f32_16x16x32_f16 (f16 in, f32 accumulate).
// Weights stay in their native [N,K] row-major layout (f16, K padded to 32)
// so both A- and B-fragments load as two contiguous b128 per lane.
// LSTM recurrence: persistent 1024-thread workgroup per direction, Whh
// fragments resident in VGPRs, h in LDS, c in accumulator registers.
// ---------------------------------------------------------------------------

#define B_ 64
#define T_ 800
#define M_ (B_ * T_)          // 51200 rows, divisible by 64
#define LOG2PI_ 1.8378770664093453f
#define LOG_HALF_ (-0.6931471805599453f)

typedef _Float16 f16;
typedef __attribute__((ext_vector_type(16))) _Float16 v16h;
typedef __attribute__((ext_vector_type(8)))  float    v8f;

union Frag16 { v16h v; f16 h[16]; uint4 q[2]; };

__device__ __forceinline__ v8f wmma16(v16h a, v16h b, v8f c) {
#if defined(__gfx1250__)
  // (neg_a, A, neg_b, B, c_mod, C, reuse_a, reuse_b)
  return __builtin_amdgcn_wmma_f32_16x16x32_f16(false, a, false, b, (short)0, c,
                                                false, false);
#else
  (void)a; (void)b;
  return c;
#endif
}

__device__ __forceinline__ float sigm_(float x) { return 1.0f / (1.0f + __expf(-x)); }

// ---------------------------------------------------------------------------
// Generic WMMA GEMM: C[M,N] = act(A16[M,K] * W16[N,K]^T + bias + bias2)
// A16: [M,K] row-major f16.  W16: [N,K] row-major f16 (native torch layout).
// block = 128 threads (4 waves); block tile 64x64; K step 32.
// B-fragment layout: lane group g holds K = g*16..g*16+15 at column n=lane%16,
// i.e. 16 contiguous K-halves of one weight row -> two b128 loads.
// act: 0 none, 1 relu, 2 sigmoid, 3 exp.  outF (f32) and/or outH (f16).
// ---------------------------------------------------------------------------
__global__ __launch_bounds__(128) void gemm_f16_kernel(
    const f16* __restrict__ A, const f16* __restrict__ W,
    const float* __restrict__ bias, const float* __restrict__ bias2,
    float* __restrict__ outF, f16* __restrict__ outH,
    int M, int N, int K, int act)
{
  __shared__ alignas(16) f16 As[64 * 32];   // [row][k]
  __shared__ alignas(16) f16 Bs[64 * 32];   // [n][k]
  const int tid   = threadIdx.x;
  const int wave  = tid >> 5;
  const int lane  = tid & 31;
  const int laneN = lane & 15;
  const int laneG = lane >> 4;
  const int rowBlk = blockIdx.x * 64;
  const int colBlk = blockIdx.y * 64;

  const v8f vzero = {0.f, 0.f, 0.f, 0.f, 0.f, 0.f, 0.f, 0.f};
  v8f acc[4];
#pragma unroll
  for (int nt = 0; nt < 4; ++nt) acc[nt] = vzero;

  const int srow = tid >> 1;        // 0..63 (A row in tile / W row in tile)
  const int scol = (tid & 1) * 16;  // 0 / 16 halves
  const int gn = colBlk + srow;     // global weight row for B staging
  const bool nvalid = (gn < N);
  const uint4 qz = {0u, 0u, 0u, 0u};

  for (int k0 = 0; k0 < K; k0 += 32) {
    const uint4* ag =
        reinterpret_cast<const uint4*>(A + (size_t)(rowBlk + srow) * K + k0 + scol);
    uint4 a0 = ag[0];
    uint4 a1 = ag[1];
    uint4 b0 = qz, b1 = qz;
    if (nvalid) {
      const uint4* bg =
          reinterpret_cast<const uint4*>(W + (size_t)gn * K + k0 + scol);
      b0 = bg[0];
      b1 = bg[1];
    }
    *reinterpret_cast<uint4*>(&As[srow * 32 + scol])     = a0;
    *reinterpret_cast<uint4*>(&As[srow * 32 + scol + 8]) = a1;
    *reinterpret_cast<uint4*>(&Bs[srow * 32 + scol])     = b0;
    *reinterpret_cast<uint4*>(&Bs[srow * 32 + scol + 8]) = b1;
    __syncthreads();

    // A fragment: 16x32 f16 (lane group selects 16 contiguous K halves)
    Frag16 a;
    const int r  = wave * 16 + laneN;
    const int kb = laneG * 16;
    a.q[0] = *reinterpret_cast<const uint4*>(&As[r * 32 + kb]);
    a.q[1] = *reinterpret_cast<const uint4*>(&As[r * 32 + kb + 8]);

#pragma unroll
    for (int nt = 0; nt < 4; ++nt) {
      Frag16 b;
      const int n  = nt * 16 + laneN;
      const int kg = laneG * 16;
      b.q[0] = *reinterpret_cast<const uint4*>(&Bs[n * 32 + kg]);
      b.q[1] = *reinterpret_cast<const uint4*>(&Bs[n * 32 + kg + 8]);
      acc[nt] = wmma16(a.v, b.v, acc[nt]);
    }
    __syncthreads();
  }

#pragma unroll
  for (int nt = 0; nt < 4; ++nt) {
    const int col = colBlk + nt * 16 + laneN;
    if (col >= N) continue;
    const float bsum = (bias ? bias[col] : 0.f) + (bias2 ? bias2[col] : 0.f);
#pragma unroll
    for (int v = 0; v < 8; ++v) {
      const int row = rowBlk + wave * 16 + v + laneG * 8;
      float xv = acc[nt][v] + bsum;
      if (act == 1)      xv = fmaxf(xv, 0.f);
      else if (act == 2) xv = 1.f / (1.f + __expf(-xv));
      else if (act == 3) xv = __expf(xv);
      if (outF) outF[(size_t)row * N + col] = xv;
      if (outH) outH[(size_t)row * N + col] = (f16)xv;
    }
  }
}

// ---------------------------------------------------------------------------
// Persistent BiLSTM recurrence. grid=(2): blockIdx 0 = fwd, 1 = bwd.
// 1024 threads = 32 waves. Wave w: M-tile m = w&3 (16 batch rows),
// hidden slice n0 = (w>>2)*16. Each wave holds i/f/g/o accumulators for its
// own (b,j) 16x16 patch -> cell state c stays in registers, activations are
// applied in WMMA accumulator layout. Whh [512,128] f16 fragments preloaded
// to VGPRs with b128 loads (K-contiguous rows).
// gpre: [M,512] f32 = x@Wih^T + bih + bhh (row = b*T + t).
// qout: [M,256] f16; fwd -> cols [0,128), bwd -> cols [128,256).
// ---------------------------------------------------------------------------
__global__ __launch_bounds__(1024) void lstm_seq_kernel(
    const float* __restrict__ gpreF, const float* __restrict__ gpreB,
    const f16* __restrict__ whhF, const f16* __restrict__ whhB,
    f16* __restrict__ qout, int T)
{
  __shared__ alignas(16) f16 hs[64 * 128];
  const int dir = blockIdx.x;
  const float* __restrict__ gpre = dir ? gpreB : gpreF;
  const f16* __restrict__ whh    = dir ? whhB : whhF;
  const int colOff = dir ? 128 : 0;

  const int tid   = threadIdx.x;
  const int wave  = tid >> 5;
  const int lane  = tid & 31;
  const int laneN = lane & 15;
  const int laneG = lane >> 4;
  const int m  = wave & 3;
  const int n0 = (wave >> 2) * 16;
  const int j  = n0 + laneN;           // hidden column 0..127

  // Preload Whh [512,128] B-fragments: [gate][k-chunk], kept in VGPRs.
  // Fragment wants rows n = gate*128 + j, K halves [kc*32+g*16, +16) -> b128 x2.
  Frag16 bf[4][4];
#pragma unroll
  for (int gb = 0; gb < 4; ++gb) {
#pragma unroll
    for (int kc = 0; kc < 4; ++kc) {
      const f16* wp = whh + (size_t)(gb * 128 + j) * 128 + kc * 32 + laneG * 16;
      bf[gb][kc].q[0] = *reinterpret_cast<const uint4*>(wp);
      bf[gb][kc].q[1] = *reinterpret_cast<const uint4*>(wp + 8);
    }
  }

  {  // h(0) = 0
    const uint4 qz = {0u, 0u, 0u, 0u};
    for (int i = tid * 8; i < 64 * 128; i += 1024 * 8)
      *reinterpret_cast<uint4*>(&hs[i]) = qz;
  }
  const v8f vzero = {0.f, 0.f, 0.f, 0.f, 0.f, 0.f, 0.f, 0.f};
  v8f cst = vzero;  // cell state, in accumulator layout
  __syncthreads();

  for (int t = 0; t < T; ++t) {
    const int tt = dir ? (T - 1 - t) : t;

    // Seed accumulators with precomputed x@Wih + bias (accumulator layout).
    v8f acc[4];
#pragma unroll
    for (int gb = 0; gb < 4; ++gb) {
#pragma unroll
      for (int v = 0; v < 8; ++v) {
        const int b = m * 16 + v + laneG * 8;
        acc[gb][v] = gpre[(size_t)(b * T + tt) * 512 + gb * 128 + j];
      }
    }
    if (t + 1 < T) {
      const int tn = dir ? (T - 2 - t) : (t + 1);
      __builtin_prefetch(&gpre[(size_t)((m * 16 + laneG * 8) * T + tn) * 512 + j], 0, 0);
    }

    // gates += h @ Whh^T  (K = 128 -> 4 WMMA per gate)
#pragma unroll
    for (int kc = 0; kc < 4; ++kc) {
      Frag16 a;
      const int r  = m * 16 + laneN;
      const int kb = kc * 32 + laneG * 16;
      a.q[0] = *reinterpret_cast<const uint4*>(&hs[r * 128 + kb]);
      a.q[1] = *reinterpret_cast<const uint4*>(&hs[r * 128 + kb + 8]);
#pragma unroll
      for (int gb = 0; gb < 4; ++gb)
        acc[gb] = wmma16(a.v, bf[gb][kc].v, acc[gb]);
    }
    __syncthreads();  // all waves done reading hs

#pragma unroll
    for (int v = 0; v < 8; ++v) {
      const float ig = sigm_(acc[0][v]);
      const float fg = sigm_(acc[1][v]);
      const float gg = tanhf(acc[2][v]);
      const float og = sigm_(acc[3][v]);
      const float cn = fg * cst[v] + ig * gg;
      cst[v] = cn;
      const float hv = og * tanhf(cn);
      const int b = m * 16 + v + laneG * 8;
      hs[b * 128 + j] = (f16)hv;
      qout[(size_t)(b * T + tt) * 256 + colOff + j] = (f16)hv;
    }
    __syncthreads();  // hs(t+1) visible to everyone
  }
}

// ---------------------------------------------------------------------------
// Small helper kernels
// ---------------------------------------------------------------------------

// src[M,K] f32 -> dst[M,Kp] f16 (zero-pad K); used for activations & weights
__global__ void cast_pad_kernel(const float* __restrict__ src,
                                f16* __restrict__ dst, int M, int K, int Kp)
{
  const int i = blockIdx.x * blockDim.x + threadIdx.x;
  if (i >= M * Kp) return;
  const int r = i / Kp;
  const int c = i - r * Kp;
  dst[i] = (c < K) ? (f16)src[(size_t)r * K + c] : (f16)0.f;
}

// feat[M,224]: cols [0,88) = pr, cols [216,224) = 0 (z cols done elsewhere)
__global__ void fill_feat_pr_kernel(const float* __restrict__ pr,
                                    f16* __restrict__ feat, int M)
{
  const int i = blockIdx.x * blockDim.x + threadIdx.x;
  if (i >= M * 224) return;
  const int r = i / 224;
  const int c = i - r * 224;
  if (c < 88)        feat[i] = (f16)pr[(size_t)r * 88 + c];
  else if (c >= 216) feat[i] = (f16)0.f;
}

// z = mu + var*eps -> f32 out + f16 into feat[:, colbase+j]
__global__ void z_combine_kernel(const float* __restrict__ mu,
                                 const float* __restrict__ var,
                                 const float* __restrict__ eps,
                                 float* __restrict__ zout,
                                 f16* __restrict__ feat, int colbase, int M)
{
  const int i = blockIdx.x * blockDim.x + threadIdx.x;
  if (i >= M * 64) return;
  const int r = i >> 6;
  const int j = i & 63;
  const float z = mu[i] + var[i] * eps[i];
  zout[i] = z;
  feat[(size_t)r * 224 + colbase + j] = (f16)z;
}

// GMM posterior over K=2 components, Z=64
__global__ void gmm_kernel(const float* __restrict__ z,
                           const float* __restrict__ mu_lk,
                           const float* __restrict__ lv_lk,
                           float* __restrict__ cls, int M)
{
  const int r = blockIdx.x * blockDim.x + threadIdx.x;
  if (r >= M) return;
  float llh[2];
#pragma unroll
  for (int k = 0; k < 2; ++k) {
    float s = 0.f;
    for (int j = 0; j < 64; ++j) {
      const float d  = z[(size_t)r * 64 + j] - mu_lk[k * 64 + j];
      const float lv = lv_lk[k * 64 + j];
      s += d * d * __expf(-lv) + lv + LOG2PI_;
    }
    llh[k] = -0.5f * s + LOG_HALF_;
  }
  const float mx  = fmaxf(llh[0], llh[1]);
  const float e0  = __expf(llh[0] - mx);
  const float e1  = __expf(llh[1] - mx);
  const float inv = 1.f / (e0 + e1);
  cls[(size_t)r * 2 + 0] = e0 * inv;
  cls[(size_t)r * 2 + 1] = e1 * inv;
}

// ---------------------------------------------------------------------------
// Host orchestration
// ---------------------------------------------------------------------------

// Input leaf order. Top level follows setup_inputs() insertion order; nested
// LSTM dicts flatten JAX-style with sorted keys ("bwd" before "fwd").
enum {
  IN_X = 0, IN_PR, IN_EPS_ART, IN_EPS_DYN, IN_CONV_W, IN_CONV_B,
  IN_ENC_B_WIH, IN_ENC_B_WHH, IN_ENC_B_BIH, IN_ENC_B_BHH,
  IN_ENC_F_WIH, IN_ENC_F_WHH, IN_ENC_F_BIH, IN_ENC_F_BHH,
  IN_MUA_W, IN_MUA_B, IN_VARA_W, IN_VARA_B,
  IN_MUD_W, IN_MUD_B, IN_VARD_W, IN_VARD_B,
  IN_MU_ART_LK, IN_MU_DYN_LK, IN_LV_ART_LK, IN_LV_DYN_LK,
  IN_D0_B_WIH, IN_D0_B_WHH, IN_D0_B_BIH, IN_D0_B_BHH,
  IN_D0_F_WIH, IN_D0_F_WHH, IN_D0_F_BIH, IN_D0_F_BHH,
  IN_D1_B_WIH, IN_D1_B_WHH, IN_D1_B_BIH, IN_D1_B_BHH,
  IN_D1_F_WIH, IN_D1_F_WHH, IN_D1_F_BIH, IN_D1_F_BHH,
  IN_OUT_W, IN_OUT_B, IN_OUT2_W, IN_OUT2_B
};

extern "C" void kernel_launch(void* const* d_in, const int* in_sizes, int n_in,
                              void* d_out, int out_size, void* d_ws, size_t ws_size,
                              hipStream_t stream) {
  (void)in_sizes; (void)n_in; (void)out_size; (void)ws_size;
  const int M = M_, T = T_;
  float* out = (float*)d_out;

  // output offsets (f32 elements), tuple order
  const size_t O_XHAT = 0;
  const size_t O_ZA   = O_XHAT + (size_t)M * 80;
  const size_t O_ACLS = O_ZA   + (size_t)M * 64;
  const size_t O_MUA  = O_ACLS + (size_t)M * 2;
  const size_t O_VARA = O_MUA  + (size_t)M * 64;
  const size_t O_ZD   = O_VARA + (size_t)M * 64;
  const size_t O_DCLS = O_ZD   + (size_t)M * 64;
  const size_t O_MUD  = O_DCLS + (size_t)M * 2;
  const size_t O_VARD = O_MUD  + (size_t)M * 64;

  // workspace carve-up
  char* ws = (char*)d_ws;
  size_t cur = 0;
  auto carve = [&](size_t bytes) -> char* {
    char* p = ws + cur;
    cur += (bytes + 255) & ~(size_t)255;
    return p;
  };
  f16* x16    = (f16*)carve((size_t)M * 96 * 2);
  f16* h16    = (f16*)carve((size_t)M * 256 * 2);
  f16* q16    = (f16*)carve((size_t)M * 256 * 2);
  f16* feat16 = (f16*)carve((size_t)M * 224 * 2);
  f16* hd0    = (f16*)carve((size_t)M * 256 * 2);
  f16* hd1    = (f16*)carve((size_t)M * 256 * 2);
  f16* hrelu  = (f16*)carve((size_t)M * 256 * 2);
  float* gF   = (float*)carve((size_t)M * 512 * 4);
  float* gB   = (float*)carve((size_t)M * 512 * 4);
  // weights, native [N, Kpad] f16
  f16* wc     = (f16*)carve((size_t)256 * 96 * 2);
  f16* wi_ef  = (f16*)carve((size_t)512 * 256 * 2);
  f16* wi_eb  = (f16*)carve((size_t)512 * 256 * 2);
  f16* wh_ef  = (f16*)carve((size_t)512 * 128 * 2);
  f16* wh_eb  = (f16*)carve((size_t)512 * 128 * 2);
  f16* wmua   = (f16*)carve((size_t)64 * 256 * 2);
  f16* wvara  = (f16*)carve((size_t)64 * 256 * 2);
  f16* wmud   = (f16*)carve((size_t)64 * 256 * 2);
  f16* wvard  = (f16*)carve((size_t)64 * 256 * 2);
  f16* wi_0f  = (f16*)carve((size_t)512 * 224 * 2);
  f16* wi_0b  = (f16*)carve((size_t)512 * 224 * 2);
  f16* wh_0f  = (f16*)carve((size_t)512 * 128 * 2);
  f16* wh_0b  = (f16*)carve((size_t)512 * 128 * 2);
  f16* wi_1f  = (f16*)carve((size_t)512 * 256 * 2);
  f16* wi_1b  = (f16*)carve((size_t)512 * 256 * 2);
  f16* wh_1f  = (f16*)carve((size_t)512 * 128 * 2);
  f16* wh_1b  = (f16*)carve((size_t)512 * 128 * 2);
  f16* wo1    = (f16*)carve((size_t)256 * 256 * 2);
  f16* wo2    = (f16*)carve((size_t)80 * 256 * 2);

  auto F = [&](int i) { return (const float*)d_in[i]; };
  // W[N,K] f32 -> W16[N,Kp] f16 (native layout, K zero-padded)
  auto castW = [&](int idx, f16* dst, int N, int K, int Kp) {
    const int n = N * Kp;
    cast_pad_kernel<<<(n + 255) / 256, 256, 0, stream>>>(F(idx), dst, N, K, Kp);
  };
  auto gemm = [&](const f16* A, const f16* W, const float* b1, const float* b2,
                  float* oF, f16* oH, int N, int K, int act) {
    dim3 grid(M / 64, (N + 63) / 64);
    gemm_f16_kernel<<<grid, 128, 0, stream>>>(A, W, b1, b2, oF, oH, M, N, K, act);
  };

  // --- weight conversion (cast + K-pad, native [N,K] layout) ---
  castW(IN_CONV_W,    wc,    256, 80,  96);
  castW(IN_ENC_F_WIH, wi_ef, 512, 256, 256);
  castW(IN_ENC_B_WIH, wi_eb, 512, 256, 256);
  castW(IN_ENC_F_WHH, wh_ef, 512, 128, 128);
  castW(IN_ENC_B_WHH, wh_eb, 512, 128, 128);
  castW(IN_MUA_W,  wmua,  64, 256, 256);
  castW(IN_VARA_W, wvara, 64, 256, 256);
  castW(IN_MUD_W,  wmud,  64, 256, 256);
  castW(IN_VARD_W, wvard, 64, 256, 256);
  castW(IN_D0_F_WIH, wi_0f, 512, 216, 224);
  castW(IN_D0_B_WIH, wi_0b, 512, 216, 224);
  castW(IN_D0_F_WHH, wh_0f, 512, 128, 128);
  castW(IN_D0_B_WHH, wh_0b, 512, 128, 128);
  castW(IN_D1_F_WIH, wi_1f, 512, 256, 256);
  castW(IN_D1_B_WIH, wi_1b, 512, 256, 256);
  castW(IN_D1_F_WHH, wh_1f, 512, 128, 128);
  castW(IN_D1_B_WHH, wh_1b, 512, 128, 128);
  castW(IN_OUT_W,  wo1, 256, 256, 256);
  castW(IN_OUT2_W, wo2, 80,  256, 256);

  // --- encoder ---
  cast_pad_kernel<<<(M * 96 + 255) / 256, 256, 0, stream>>>(F(IN_X), x16, M, 80, 96);
  gemm(x16, wc, F(IN_CONV_B), nullptr, nullptr, h16, 256, 96, /*relu*/1);

  gemm(h16, wi_ef, F(IN_ENC_F_BIH), F(IN_ENC_F_BHH), gF, nullptr, 512, 256, 0);
  gemm(h16, wi_eb, F(IN_ENC_B_BIH), F(IN_ENC_B_BHH), gB, nullptr, 512, 256, 0);
  lstm_seq_kernel<<<2, 1024, 0, stream>>>(gF, gB, wh_ef, wh_eb, q16, T);

  // --- latent heads ---
  gemm(q16, wmua,  F(IN_MUA_B),  nullptr, out + O_MUA,  nullptr, 64, 256, 0);
  gemm(q16, wvara, F(IN_VARA_B), nullptr, out + O_VARA, nullptr, 64, 256, /*exp*/3);
  gemm(q16, wmud,  F(IN_MUD_B),  nullptr, out + O_MUD,  nullptr, 64, 256, 0);
  gemm(q16, wvard, F(IN_VARD_B), nullptr, out + O_VARD, nullptr, 64, 256, /*exp*/3);

  fill_feat_pr_kernel<<<(M * 224 + 255) / 256, 256, 0, stream>>>(F(IN_PR), feat16, M);
  z_combine_kernel<<<(M * 64 + 255) / 256, 256, 0, stream>>>(
      out + O_MUA, out + O_VARA, F(IN_EPS_ART), out + O_ZA, feat16, 88, M);
  z_combine_kernel<<<(M * 64 + 255) / 256, 256, 0, stream>>>(
      out + O_MUD, out + O_VARD, F(IN_EPS_DYN), out + O_ZD, feat16, 152, M);
  gmm_kernel<<<(M + 255) / 256, 256, 0, stream>>>(
      out + O_ZA, F(IN_MU_ART_LK), F(IN_LV_ART_LK), out + O_ACLS, M);
  gmm_kernel<<<(M + 255) / 256, 256, 0, stream>>>(
      out + O_ZD, F(IN_MU_DYN_LK), F(IN_LV_DYN_LK), out + O_DCLS, M);

  // --- decoder ---
  gemm(feat16, wi_0f, F(IN_D0_F_BIH), F(IN_D0_F_BHH), gF, nullptr, 512, 224, 0);
  gemm(feat16, wi_0b, F(IN_D0_B_BIH), F(IN_D0_B_BHH), gB, nullptr, 512, 224, 0);
  lstm_seq_kernel<<<2, 1024, 0, stream>>>(gF, gB, wh_0f, wh_0b, hd0, T);

  gemm(hd0, wi_1f, F(IN_D1_F_BIH), F(IN_D1_F_BHH), gF, nullptr, 512, 256, 0);
  gemm(hd0, wi_1b, F(IN_D1_B_BIH), F(IN_D1_B_BHH), gB, nullptr, 512, 256, 0);
  lstm_seq_kernel<<<2, 1024, 0, stream>>>(gF, gB, wh_1f, wh_1b, hd1, T);

  gemm(hd1,   wo1, F(IN_OUT_B),  nullptr, nullptr,      hrelu, 256, 256, /*relu*/1);
  gemm(hrelu, wo2, F(IN_OUT2_B), nullptr, out + O_XHAT, nullptr, 80, 256, /*sigmoid*/2);
}
